// Attention_71554155151927
// MI455X (gfx1250) — matrix operand order repaired and checked by reading, not compiled, not run on hardware
//
#include <hip/hip_runtime.h>
#include <hip/hip_bf16.h>

// MI455X / gfx1250 attention (causal) — bf16 WMMA pipeline, f32 accumulate.
// Phase 1: QKV projections -> bf16 workspace (V stored transposed).
// Phase 2: flash-attention per 128-row query tile, 8 waves/block.
//          K/V^T chunks double-buffered in LDS by the Tensor Data Mover
//          (tensor_load_to_lds, TENSORcnt) with hardware-inserted LDS padding
//          for bank-conflict-free fragment reads; DMA of chunk c+1 overlaps
//          the 32 WMMAs + online softmax of chunk c.

typedef __attribute__((ext_vector_type(16))) __bf16    v16bf;
typedef __attribute__((ext_vector_type(8)))  float     v8f;
typedef __attribute__((ext_vector_type(4)))  unsigned  v4u;
typedef __attribute__((ext_vector_type(8)))  unsigned  v8u;

union FragBF {            // 16 bf16 = 8 VGPRs
    v16bf v;
    unsigned u[8];
};

#define EMB 256
#define SEQ 4096
#define NB  4

// LDS strides in halves (match TDM pad settings; bank-conflict-free)
#define KSTR 264          // 528 B per 32-key K row   (512 B data + 16 B pad)
#define VSTR 40           //  80 B per V^T row        ( 64 B data + 16 B pad)
#define PSTR 40           //  80 B per P row

// native bf16 conversion (lowers to v_cvt_pk_bf16_f32 on gfx1250)
static __device__ __forceinline__ unsigned short f2bfu(float f) {
    union { __bf16 h; unsigned short u; } c;
    c.h = (__bf16)f;
    return c.u;
}
static __device__ __forceinline__ unsigned pack2bf(float lo, float hi) {
    union { __bf16 h[2]; unsigned u; } c;
    c.h[0] = (__bf16)lo;
    c.h[1] = (__bf16)hi;
    return c.u;
}
// A-matrix (16xK, 16-bit) K-offset for VGPR j, lane-half hi (contiguous: k = 16*hi + 2*j)
static __device__ __forceinline__ int koffA(int j, int hi) {
    return hi * 16 + 2 * j;
}

// ---- Tensor DMA descriptor build (ISA 8.3/8.4) + issue -------------------
// group0: [1:0]=count(1), [63:32]=lds_addr, [120:64]=global_addr, [127:126]=2
// group1: [17:16]=data_size(1=2B), [20]=pad_en, [24:22]=pad_interval,
//         [31:25]=pad_amount, [79:48]=tensor_dim0, [111:80]=tensor_dim1,
//         [127:112]=tile_dim0, [143:128]=tile_dim1, [207:160]=dim0_stride
static __device__ __forceinline__ v8u tdm_group1(unsigned td0, unsigned td1,
                                                 unsigned tile0, unsigned tile1,
                                                 unsigned long long st0,
                                                 unsigned padi, unsigned pada) {
    v8u g;
    g.s0 = (1u << 16) | (1u << 20) | (padi << 22) | (pada << 25);
    g.s1 = (td0 & 0xFFFFu) << 16;
    g.s2 = (td0 >> 16) | ((td1 & 0xFFFFu) << 16);
    g.s3 = (td1 >> 16) | ((tile0 & 0xFFFFu) << 16);
    g.s4 = tile1 & 0xFFFFu;                       // tile_dim2 = 0
    g.s5 = (unsigned)(st0 & 0xFFFFFFFFu);
    g.s6 = (unsigned)(st0 >> 32);                 // dim1_stride = 0
    g.s7 = 0u;
    return g;
}
static __device__ __forceinline__ v4u tdm_group0(unsigned lds_addr,
                                                 unsigned long long gaddr) {
    v4u g;
    g.s0 = 1u;                                    // count = 1, no gather
    g.s1 = lds_addr;
    g.s2 = (unsigned)(gaddr & 0xFFFFFFFFu);
    g.s3 = (unsigned)((gaddr >> 32) & 0x01FFFFFFu) | (2u << 30);  // type=2
    return g;
}
static __device__ __forceinline__ void tdm_load(const v4u& g0, const v8u& g1) {
    asm volatile("tensor_load_to_lds %0, %1" :: "s"(g0), "s"(g1) : "memory");
}

// ---------------------------------------------------------------------------
// Kernel 1: y = x @ W^T + b  for Q/K/V.  grid = (BS/16, E/16, 3), block = 32.
// z==0: Q from X (row-major bf16 out), z==1: K from Z, z==2: V from Z
// stored transposed [f][s].
// ---------------------------------------------------------------------------
__global__ __launch_bounds__(32)
void qkv_proj_kernel(const float* __restrict__ X, const float* __restrict__ Z,
                     const float* __restrict__ Wq, const float* __restrict__ bq,
                     const float* __restrict__ Wk, const float* __restrict__ bk,
                     const float* __restrict__ Wv, const float* __restrict__ bv,
                     unsigned short* __restrict__ qb,
                     unsigned short* __restrict__ kb,
                     unsigned short* __restrict__ vtb) {
    const int lane = threadIdx.x;
    const int m    = lane & 15;
    const int hi   = lane >> 4;
    const int mbase = blockIdx.x * 16;          // global row (b*S + s)
    const int f0    = blockIdx.y * 16;          // output feature tile
    const int which = blockIdx.z;

    const float* inp  = (which == 0) ? X : Z;
    const float* W    = (which == 0) ? Wq : ((which == 1) ? Wk : Wv);
    const float* bias = (which == 0) ? bq : ((which == 1) ? bk : bv);

    const size_t arow = (size_t)(mbase + m) * EMB;
    const size_t wrow = (size_t)(f0 + m) * EMB;     // lane's B column = W row f

    v8f c = {};
    #pragma unroll
    for (int ec = 0; ec < 8; ++ec) {
        const int e0 = ec * 32;
        FragBF a, b;
        #pragma unroll
        for (int j = 0; j < 8; ++j) {
            const int ka = e0 + koffA(j, hi);
            float2 xa = *(const float2*)(inp + arow + ka);
            a.u[j] = pack2bf(xa.x, xa.y);
            const int kbk = e0 + hi * 16 + 2 * j;
            float2 xw = *(const float2*)(W + wrow + kbk);
            b.u[j] = pack2bf(xw.x, xw.y);
        }
        c = __builtin_amdgcn_wmma_f32_16x16x32_bf16(false, a.v, false, b.v,
                                                    (short)0, c, false, false);
    }

    const float bval = bias[f0 + m];            // n = lane&15, same for all rows
    #pragma unroll
    for (int r = 0; r < 8; ++r) {
        const int   srow = mbase + r + 8 * hi;  // D row M = r + 8*hi
        const int   f    = f0 + m;              // D col N = lane&15
        const unsigned short h = f2bfu(c[r] + bval);
        if (which == 2) {
            const int bb = srow >> 12;          // / SEQ
            const int ss = srow & (SEQ - 1);
            vtb[((size_t)bb * EMB + f) * SEQ + ss] = h;
        } else {
            unsigned short* dst = which ? kb : qb;
            dst[(size_t)srow * EMB + f] = h;
        }
    }
}

// ---------------------------------------------------------------------------
// Kernel 2: causal flash attention. grid = B*(S/128), block = 256 (8 waves).
// ---------------------------------------------------------------------------
__global__ __launch_bounds__(256)
void flash_attn_kernel(const unsigned short* __restrict__ qb,
                       const unsigned short* __restrict__ kb,
                       const unsigned short* __restrict__ vtb,
                       float* __restrict__ out) {
    __shared__ unsigned short kbuf[2][32 * KSTR];   // double-buffered K tiles
    __shared__ unsigned short vbuf[2][EMB * VSTR];  // double-buffered V^T tiles
    __shared__ unsigned short pbuf[8][16 * PSTR];   // per-wave P tile (padded)

    const int tid  = threadIdx.x;
    const int wave = tid >> 5;
    const int lane = tid & 31;
    const int m    = lane & 15;
    const int hi   = lane >> 4;

    const int tile = blockIdx.x;
    const int bidx = tile >> 5;                 // / (S/128)
    const int qt   = tile & 31;
    const int m0   = qt * 128 + wave * 16;      // this wave's 16 query rows

    const unsigned kbase0 = (unsigned)(size_t)&kbuf[0][0];
    const unsigned kbase1 = (unsigned)(size_t)&kbuf[1][0];
    const unsigned vbase0 = (unsigned)(size_t)&vbuf[0][0];
    const unsigned vbase1 = (unsigned)(size_t)&vbuf[1][0];

    // ---- preload Q fragments (16 rows x 256, 8 chunks of K=32) ----
    FragBF qf[8];
    {
        const unsigned short* qrow =
            qb + ((size_t)bidx * SEQ + m0 + m) * EMB;
        #pragma unroll
        for (int ec = 0; ec < 8; ++ec)
            #pragma unroll
            for (int j = 0; j < 8; ++j)
                qf[ec].u[j] = *(const unsigned*)(qrow + ec * 32 + koffA(j, hi));
    }

    v8f acc[16];
    #pragma unroll
    for (int t = 0; t < 16; ++t) acc[t] = (v8f){};
    float rmax[8], rsum[8];
    #pragma unroll
    for (int r = 0; r < 8; ++r) { rmax[r] = -3.0e38f; rsum[r] = 0.0f; }

    // TDM descriptors (uniform; only the global address moves per chunk)
    const unsigned long long kga0 =
        (unsigned long long)(kb + (size_t)bidx * SEQ * EMB);
    const unsigned long long vga0 =
        (unsigned long long)(vtb + (size_t)bidx * EMB * SEQ);
    // K: 32 rows x 256 bf16, row 512B, pad 16B every 512B (interval 2^(6+3))
    const v8u kg1 = tdm_group1(EMB, 1u << 20, EMB, 32, EMB, 6u, 3u);
    // V^T: 256 rows x 32 bf16, row 64B, pad 16B every 64B (interval 2^(3+3))
    const v8u vg1 = tdm_group1(SEQ, 1u << 20, 32, EMB, SEQ, 3u, 3u);

    const int nchunks = qt * 4 + 4;             // only j0 <= q_max (causal)

    // prologue: stage chunk 0 into buffer set 0
    if (wave == 0) {
        tdm_load(tdm_group0(kbase0, kga0), kg1);
        tdm_load(tdm_group0(vbase0, vga0), vg1);
    }

    for (int c32 = 0; c32 < nchunks; ++c32) {
        const int j0 = c32 * 32;
        const int p  = c32 & 1;
        const unsigned short* kcur = kbuf[p];
        const unsigned short* vcur = vbuf[p];

        // publish chunk c32; overlap DMA of chunk c32+1 with compute
        if (wave == 0) __builtin_amdgcn_s_wait_tensorcnt((short)0);
        __syncthreads();
        if (wave == 0 && c32 + 1 < nchunks) {
            const unsigned long long off =
                (unsigned long long)(j0 + 32) * 2;      // bytes along keys
            tdm_load(tdm_group0(p ? kbase0 : kbase1, kga0 + off * EMB), kg1);
            tdm_load(tdm_group0(p ? vbase0 : vbase1, vga0 + off), vg1);
        }

        // ---- scores S = Q K^T (16 x 32 per wave) ----
        v8f s[2];
        #pragma unroll
        for (int ns = 0; ns < 2; ++ns) {
            v8f sc = {};
            #pragma unroll
            for (int ec = 0; ec < 8; ++ec) {
                FragBF kf;
                const unsigned short* kr = kcur + (ns * 16 + m) * KSTR
                                                + ec * 32 + hi * 16;
                #pragma unroll
                for (int j = 0; j < 8; ++j)
                    kf.u[j] = *(const unsigned*)(kr + 2 * j);
                sc = __builtin_amdgcn_wmma_f32_16x16x32_bf16(
                         false, qf[ec].v, false, kf.v, (short)0, sc, false, false);
            }
            s[ns] = sc;
        }

        // ---- scale + causal mask ----
        #pragma unroll
        for (int ns = 0; ns < 2; ++ns)
            #pragma unroll
            for (int r = 0; r < 8; ++r) {
                const int qg = m0 + r + 8 * hi;
                const int kg = j0 + ns * 16 + m;
                float v = s[ns][r] * 0.0625f;   // 1/sqrt(256)
                if (kg > qg) v = -3.0e38f;
                s[ns][r] = v;
            }

        // ---- online softmax (row = 16 lanes of one half-wave) ----
        float scl[8];
        #pragma unroll
        for (int r = 0; r < 8; ++r) {
            float mx = fmaxf(s[0][r], s[1][r]);
            #pragma unroll
            for (int off = 8; off >= 1; off >>= 1)
                mx = fmaxf(mx, __shfl_xor(mx, off, 32));
            const float nm = fmaxf(rmax[r], mx);
            const float p0 = __expf(s[0][r] - nm);
            const float p1 = __expf(s[1][r] - nm);
            const float rs = __expf(rmax[r] - nm);
            float ps = p0 + p1;
            #pragma unroll
            for (int off = 8; off >= 1; off >>= 1)
                ps += __shfl_xor(ps, off, 32);
            rsum[r] = rsum[r] * rs + ps;
            rmax[r] = nm;
            scl[r]  = rs;
            pbuf[wave][(r + 8 * hi) * PSTR + m]      = f2bfu(p0);
            pbuf[wave][(r + 8 * hi) * PSTR + 16 + m] = f2bfu(p1);
        }
        #pragma unroll
        for (int t = 0; t < 16; ++t)
            #pragma unroll
            for (int r = 0; r < 8; ++r) acc[t][r] *= scl[r];

        // same-wave LDS RAW: DS ops are in-order per wave; fence the compiler
        asm volatile("s_wait_dscnt 0x0" ::: "memory");

        // ---- P (D-layout) re-read as A fragment ----
        FragBF pf;
        #pragma unroll
        for (int j = 0; j < 8; ++j)
            pf.u[j] = *(const unsigned*)&pbuf[wave][m * PSTR + koffA(j, hi)];

        // ---- O += P @ V  (16 x 256) ----
        #pragma unroll
        for (int ft = 0; ft < 16; ++ft) {
            FragBF vf;
            const unsigned short* vr = vcur + (ft * 16 + m) * VSTR + hi * 16;
            #pragma unroll
            for (int j = 0; j < 8; ++j)
                vf.u[j] = *(const unsigned*)(vr + 2 * j);
            acc[ft] = __builtin_amdgcn_wmma_f32_16x16x32_bf16(
                          false, pf.v, false, vf.v, (short)0, acc[ft],
                          false, false);
        }
    }

    // ---- epilogue: normalize and store f32 ----
    #pragma unroll
    for (int ft = 0; ft < 16; ++ft)
        #pragma unroll
        for (int r = 0; r < 8; ++r) {
            const int row = m0 + r + 8 * hi;
            out[((size_t)bidx * SEQ + row) * EMB + ft * 16 + m] =
                acc[ft][r] / rsum[r];
        }
}

// ---------------------------------------------------------------------------
extern "C" void kernel_launch(void* const* d_in, const int* in_sizes, int n_in,
                              void* d_out, int out_size, void* d_ws, size_t ws_size,
                              hipStream_t stream) {
    const float* X  = (const float*)d_in[0];
    const float* Z  = (const float*)d_in[1];
    // d_in[2] = mask (causal tril) — enforced analytically, never read
    const float* Wq = (const float*)d_in[3];
    const float* bq = (const float*)d_in[4];
    const float* Wk = (const float*)d_in[5];
    const float* bk = (const float*)d_in[6];
    const float* Wv = (const float*)d_in[7];
    const float* bv = (const float*)d_in[8];
    float* out = (float*)d_out;

    const size_t nElem = (size_t)NB * SEQ * EMB;          // 4M elements
    unsigned short* qb  = (unsigned short*)d_ws;          //  8 MB bf16 Q
    unsigned short* kb  = qb  + nElem;                    //  8 MB bf16 K
    unsigned short* vtb = kb  + nElem;                    //  8 MB bf16 V^T
    (void)ws_size; (void)in_sizes; (void)n_in; (void)out_size;

    dim3 g1(NB * SEQ / 16, EMB / 16, 3);
    qkv_proj_kernel<<<g1, 32, 0, stream>>>(X, Z, Wq, bq, Wk, bk, Wv, bv,
                                           qb, kb, vtb);

    dim3 g2(NB * (SEQ / 128));
    flash_attn_kernel<<<g2, 256, 0, stream>>>(qb, kb, vtb, out);
}
